// corr_net_81930796139053
// MI455X (gfx1250) — matrix operand order repaired and checked by reading, not compiled, not run on hardware
//
#include <hip/hip_runtime.h>
#include <hip/hip_bf16.h>
#include <stdint.h>

// ---------------------------------------------------------------------------
// CDNA5 (gfx1250) correlation-network pipeline.
//  - bf16 WMMA (v_wmma_f32_16x16x32_bf16) for all GEMM-shaped work
//  - double-buffered async global->LDS staging (global_load_async_to_lds_b128,
//    ASYNCcnt) overlapping DMA with WMMA compute, one barrier per K step
//  - VALU for the (89% zero) cost volume, exploiting padding structure
// Workspace usage ~161 MB (X region aliased by the padded corr volume).
// ---------------------------------------------------------------------------

typedef __attribute__((ext_vector_type(16))) __bf16 v16bf;
typedef __attribute__((ext_vector_type(8)))  float  v8f;

union FragBF { v16bf v; uint4 q[2]; };

__device__ __forceinline__ unsigned short f2bf(float f) {
  unsigned int u = __float_as_uint(f);
  u += 0x7FFFu + ((u >> 16) & 1u);           // round-to-nearest-even
  return (unsigned short)(u >> 16);
}
__device__ __forceinline__ float bf2f(unsigned short h) {
  return __uint_as_float(((unsigned int)h) << 16);
}

#define Bb 128
#define Cc 512
#define Hh 10
#define Ww 20
#define HW 200
#define NPIX (Bb*HW)            // 25600
#define CORR_CH 1323
#define CORR_PAD 1344           // 42*32
#define CAT_CH 1536             // 48*32

// ---------------------------------------------------------------------------
// K0a: feats [B,512,200] f32 -> X [B,200,512] bf16 (channels-last transpose)
// ---------------------------------------------------------------------------
__global__ void k0_feat(const float* __restrict__ f1, const float* __restrict__ f2,
                        const float* __restrict__ f3,
                        unsigned short* __restrict__ x1, unsigned short* __restrict__ x2,
                        unsigned short* __restrict__ x3)
{
  const float* src = blockIdx.z == 0 ? f1 : (blockIdx.z == 1 ? f2 : f3);
  unsigned short* dst = blockIdx.z == 0 ? x1 : (blockIdx.z == 1 ? x2 : x3);
  const size_t n = (size_t)Bb * Cc * HW;
  for (size_t i = (size_t)blockIdx.x * blockDim.x + threadIdx.x; i < n;
       i += (size_t)gridDim.x * blockDim.x) {
    size_t b = i / (Cc * HW);
    size_t r = i % (Cc * HW);
    size_t c = r / HW;
    size_t hw = r % HW;
    dst[(b * HW + hw) * Cc + c] = f2bf(src[i]);
  }
}

// ---------------------------------------------------------------------------
// K0b: 512x512 weights f32 -> bf16 (layout already [N=o][K=c])
// ---------------------------------------------------------------------------
__global__ void k0_w512(const float* __restrict__ a, const float* __restrict__ b,
                        const float* __restrict__ c,
                        unsigned short* __restrict__ da, unsigned short* __restrict__ db,
                        unsigned short* __restrict__ dc)
{
  const float* src = blockIdx.z == 0 ? a : (blockIdx.z == 1 ? b : c);
  unsigned short* dst = blockIdx.z == 0 ? da : (blockIdx.z == 1 ? db : dc);
  const size_t n = (size_t)Cc * Cc;
  for (size_t i = (size_t)blockIdx.x * blockDim.x + threadIdx.x; i < n;
       i += (size_t)gridDim.x * blockDim.x)
    dst[i] = f2bf(src[i]);
}

// ---------------------------------------------------------------------------
// K0c: 3x3 conv weight [3][CI][3][3] f32 -> BT bf16 [16][9*CIpad]
//      k index = (ky*3+kx)*CIpad + ci ; rows 3..15 and ci>=CI are zero.
// ---------------------------------------------------------------------------
__global__ void k0_wconv(const float* __restrict__ src, unsigned short* __restrict__ dst,
                         int CI, int CIpad)
{
  const size_t n = (size_t)16 * 9 * CIpad;
  for (size_t i = (size_t)blockIdx.x * blockDim.x + threadIdx.x; i < n;
       i += (size_t)gridDim.x * blockDim.x) {
    int nrow = (int)(i / (9 * (size_t)CIpad));
    int rem  = (int)(i % (9 * (size_t)CIpad));
    int tap = rem / CIpad;
    int ci  = rem % CIpad;
    float v = 0.f;
    if (nrow < 3 && ci < CI) {
      int ky = tap / 3, kx = tap % 3;
      v = src[(((size_t)nrow * CI + ci) * 3 + ky) * 3 + kx];
    }
    dst[i] = f2bf(v);
  }
}

// ---------------------------------------------------------------------------
// K1: 1x1 conv = GEMM  out[pix, o] = X[pix, c] * W[o, c] + bias[o]
//     bf16 WMMA, block tile 64x64, 8 waves (4 along M x 2 along N).
//     Ping-pong LDS buffers fed by CDNA5 async global->LDS DMA: each
//     iteration waits only on its own ASYNCcnt, one workgroup barrier,
//     then issues the NEXT tile's DMA so it overlaps the WMMA work.
// ---------------------------------------------------------------------------
__global__ __launch_bounds__(256) void k1_conv1x1(
    const unsigned short* __restrict__ Xa, const unsigned short* __restrict__ Xb,
    const unsigned short* __restrict__ Xc,
    const unsigned short* __restrict__ Wa, const unsigned short* __restrict__ Wb,
    const unsigned short* __restrict__ Wc,
    const float* __restrict__ ba, const float* __restrict__ bbv,
    const float* __restrict__ bc,
    unsigned short* __restrict__ fcat)
{
  const int conv = blockIdx.z;
  const unsigned short* X = conv == 0 ? Xa : (conv == 1 ? Xb : Xc);
  const unsigned short* W = conv == 0 ? Wa : (conv == 1 ? Wb : Wc);
  const float* bias = conv == 0 ? ba : (conv == 1 ? bbv : bc);
  const int m0 = blockIdx.x * 64;     // pixel-row tile
  const int n0 = blockIdx.y * 64;     // out-channel tile

  __shared__ unsigned short As[2][64][32];   // ping-pong
  __shared__ unsigned short Bs[2][64][32];

  const int tid  = threadIdx.x;
  const int wave = tid >> 5;
  const int lane = tid & 31;
  const int wm = wave & 3;            // wave M index 0..3
  const int wn = wave >> 2;           // wave N index 0..1
  const int lm = lane & 15;
  const int kh = lane >> 4;

  // per-thread staging addresses; buffer stride = 64*32*2 = 4096 bytes
  const int r = tid >> 2, p = tid & 3;                // 64 rows x 4 chunks
  const unsigned lds_a = (unsigned)(uintptr_t)(&As[0][r][0]) + (unsigned)(p * 16);
  const unsigned lds_b = (unsigned)(uintptr_t)(&Bs[0][r][0]) + (unsigned)(p * 16);
  const unsigned short* ga = X + (size_t)(m0 + r) * Cc + p * 8;
  const unsigned short* gb = W + (size_t)(n0 + r) * Cc + p * 8;

  v8f acc0 = {}; v8f acc1 = {};

  // prologue: DMA first K tile into buffer 0
  asm volatile("global_load_async_to_lds_b128 %0, %1, off"
               :: "v"(lds_a), "v"(ga) : "memory");
  asm volatile("global_load_async_to_lds_b128 %0, %1, off"
               :: "v"(lds_b), "v"(gb) : "memory");

  for (int k0 = 0; k0 < Cc; k0 += 32) {
    const int cur = (k0 >> 5) & 1;
    const int nxt = cur ^ 1;
    // my DMA for 'cur' done; barrier -> tile visible to all waves, and all
    // waves have finished their previous-iteration reads of buffer 'nxt'
    asm volatile("s_wait_asynccnt 0" ::: "memory");
    __syncthreads();
    // issue next tile's DMA into the other buffer; overlaps with the WMMAs
    if (k0 + 32 < Cc) {
      asm volatile("global_load_async_to_lds_b128 %0, %1, off"
                   :: "v"(lds_a + (unsigned)(nxt << 12)), "v"(ga + k0 + 32)
                   : "memory");
      asm volatile("global_load_async_to_lds_b128 %0, %1, off"
                   :: "v"(lds_b + (unsigned)(nxt << 12)), "v"(gb + k0 + 32)
                   : "memory");
    }

    FragBF a, b0, b1;
    a.q[0]  = *(const uint4*)&As[cur][wm * 16 + lm][kh * 16];
    a.q[1]  = *(const uint4*)&As[cur][wm * 16 + lm][kh * 16 + 8];
    b0.q[0] = *(const uint4*)&Bs[cur][wn * 32 + lm][kh * 16];
    b0.q[1] = *(const uint4*)&Bs[cur][wn * 32 + lm][kh * 16 + 8];
    b1.q[0] = *(const uint4*)&Bs[cur][wn * 32 + 16 + lm][kh * 16];
    b1.q[1] = *(const uint4*)&Bs[cur][wn * 32 + 16 + lm][kh * 16 + 8];
    acc0 = __builtin_amdgcn_wmma_f32_16x16x32_bf16(false, a.v, false, b0.v,
                                                   (short)0, acc0, false, false);
    acc1 = __builtin_amdgcn_wmma_f32_16x16x32_bf16(false, a.v, false, b1.v,
                                                   (short)0, acc1, false, false);
  }

  // D layout: lane holds D[m = r + 8*(lane/16)][n = lane%16]
  #pragma unroll
  for (int rr = 0; rr < 8; ++rr) {
    int m = wm * 16 + rr + 8 * kh;
    size_t pix = (size_t)m0 + m;
    int na = n0 + wn * 32 + lm;
    int nb = na + 16;
    unsigned short* dst = fcat + pix * CAT_CH + conv * Cc;
    dst[na] = f2bf(acc0[rr] + bias[na]);
    dst[nb] = f2bf(acc1[rr] + bias[nb]);
  }
}

// ---------------------------------------------------------------------------
// K2: cost volume. For pixel (y,x) only displacements whose target pixel is
//     in-bounds are nonzero: rows r with parity(y) (5 of them), cols c with
//     parity(x) (10 of them). Everything else stays 0 (memset). Fused
//     leaky-relu(0.1). Block = (y, pair, b); C processed in 4 chunks of 128.
// ---------------------------------------------------------------------------
__global__ __launch_bounds__(256) void k2_corr(const unsigned short* __restrict__ fcat,
                                               unsigned short* __restrict__ corrP)
{
  const int y = blockIdx.x, pair = blockIdx.y, b = blockIdx.z;
  const int aoff = (pair == 2) ? 512 : 0;           // (f1,f2),(f1,f3),(f2,f3)
  const int boff = (pair == 0) ? 512 : 1024;
  const int tid = threadIdx.x;
  const int yp = y & 1;

  __shared__ unsigned int f1s[20 * 64];             // 20 x * 128ch (as 64 uints)
  __shared__ unsigned int f2s[5 * 20 * 64];         // 5 rows * 20 cols * 128ch

  float acc[4] = {0.f, 0.f, 0.f, 0.f};

  for (int ck = 0; ck < 4; ++ck) {
    __syncthreads();
    for (int j = tid; j < 20 * 64; j += 256) {
      int x = j >> 6, e = j & 63;
      const unsigned int* s = (const unsigned int*)
          (fcat + ((size_t)(b * HW + y * Ww + x)) * CAT_CH + aoff + ck * 128);
      f1s[j] = s[e];
    }
    for (int j = tid; j < 5 * 20 * 64; j += 256) {
      int rr = j / (20 * 64);
      int rem = j % (20 * 64);
      int x2 = rem >> 6, e = rem & 63;
      int r = yp + 2 * rr;
      const unsigned int* s = (const unsigned int*)
          (fcat + ((size_t)(b * HW + r * Ww + x2)) * CAT_CH + boff + ck * 128);
      f2s[j] = s[e];
    }
    __syncthreads();
    #pragma unroll
    for (int i = 0; i < 4; ++i) {
      int o = tid + 256 * i;
      if (o >= 1000) break;                         // 20x * 5r * 10c outputs
      int x = o / 50, rr = (o % 50) / 10, ci = o % 10;
      int c = (x & 1) + 2 * ci;
      const unsigned int* u1 = &f1s[x * 64];
      const unsigned int* u2 = &f2s[(rr * 20 + c) * 64];
      float s = acc[i];
      for (int e = 0; e < 64; ++e) {
        unsigned int av = u1[e], bv = u2[e];
        s += bf2f((unsigned short)(av & 0xFFFFu)) * bf2f((unsigned short)(bv & 0xFFFFu));
        s += bf2f((unsigned short)(av >> 16))     * bf2f((unsigned short)(bv >> 16));
      }
      acc[i] = s;
    }
  }
  #pragma unroll
  for (int i = 0; i < 4; ++i) {
    int o = tid + 256 * i;
    if (o >= 1000) break;
    int x = o / 50, rr = (o % 50) / 10, ci = o % 10;
    int dyi = 10 - (y >> 1) + rr;                   // = (r - y + 20)/2
    int dxi = 10 - (x >> 1) + ci;                   // = (c - x + 20)/2
    int ch = pair * 441 + dyi * 21 + dxi;
    float v = acc[i] * (1.0f / 512.0f);
    v = (v < 0.f) ? 0.1f * v : v;                   // leaky relu
    corrP[((size_t)(b * HW + y * Ww + x)) * CORR_PAD + ch] = f2bf(v);
  }
}

// ---------------------------------------------------------------------------
// K3/K4: 3x3 conv head as implicit GEMM. M = 16 pixels per wave, N = 16
//        (3 used), K = 9 taps * chPad. Output fp32 [B][o*200 + pix], fused
//        bias (+optional relu) = exactly the reference reshape layout.
// ---------------------------------------------------------------------------
__global__ __launch_bounds__(256) void k34_conv3x3(
    const unsigned short* __restrict__ src, int chPad,
    const unsigned short* __restrict__ BT,      // [16][9*chPad] bf16
    const float* __restrict__ bias, int relu,
    float* __restrict__ out600)
{
  const int tid = threadIdx.x;
  const int wave = tid >> 5, lane = tid & 31;
  const int lm = lane & 15, kh = lane >> 4;
  const int tile = blockIdx.x * 8 + wave;       // 1600 tiles of 16 pixels
  const int pm = tile * 16 + lm;                // this lane's A-row pixel
  const int b = pm / HW, rem = pm % HW, y = rem / Ww, x = rem % Ww;

  const unsigned short* base[9];
  #pragma unroll
  for (int ky = 0; ky < 3; ++ky)
    #pragma unroll
    for (int kx = 0; kx < 3; ++kx) {
      int yy = y + ky - 1, xx = x + kx - 1;
      bool ok = (yy >= 0 && yy < Hh && xx >= 0 && xx < Ww);
      base[ky * 3 + kx] = ok ? (src + ((size_t)(b * HW + yy * Ww + xx)) * chPad)
                             : nullptr;
    }
  const int stepsPerTap = chPad / 32;
  const unsigned short* brow = BT + (size_t)lm * (9 * (size_t)chPad);

  v8f acc = {};
  #pragma unroll
  for (int tap = 0; tap < 9; ++tap) {
    const unsigned short* ab = base[tap];
    const unsigned short* bp = brow + (size_t)tap * chPad;
    for (int s = 0; s < stepsPerTap; ++s) {
      int cc0 = s * 32;
      FragBF a, bf;
      if (ab) {
        a.q[0] = *(const uint4*)(ab + cc0 + kh * 16);
        a.q[1] = *(const uint4*)(ab + cc0 + kh * 16 + 8);
      } else {
        a.q[0] = make_uint4(0u, 0u, 0u, 0u);
        a.q[1] = make_uint4(0u, 0u, 0u, 0u);
      }
      bf.q[0] = *(const uint4*)(bp + cc0 + kh * 16);
      bf.q[1] = *(const uint4*)(bp + cc0 + kh * 16 + 8);
      acc = __builtin_amdgcn_wmma_f32_16x16x32_bf16(false, a.v, false, bf.v,
                                                    (short)0, acc, false, false);
    }
  }

  if (lm < 3) {
    const int n = lm;
    #pragma unroll
    for (int rr = 0; rr < 8; ++rr) {
      int m = rr + 8 * kh;
      int pix = tile * 16 + m;
      int bo = pix / HW, pp = pix % HW;
      float v = acc[rr] + bias[n];
      if (relu) v = fmaxf(v, 0.f);
      out600[(size_t)bo * 600 + n * HW + pp] = v;
    }
  }
}

// ---------------------------------------------------------------------------
// K5: per-batch MLP chains (600->256->128 twice) + final 256->2. One block/b.
// ---------------------------------------------------------------------------
__global__ __launch_bounds__(256) void k5_mlp(
    const float* __restrict__ inA, const float* __restrict__ inB,
    const float* __restrict__ w1a, const float* __restrict__ b1a,
    const float* __restrict__ w2a, const float* __restrict__ b2a,
    const float* __restrict__ w1b, const float* __restrict__ b1b,
    const float* __restrict__ w2b, const float* __restrict__ b2b,
    const float* __restrict__ wo,  const float* __restrict__ bo,
    float* __restrict__ out)
{
  const int b = blockIdx.x;
  const int tid = threadIdx.x;
  __shared__ float sbuf[600];
  __shared__ float h1[256];
  __shared__ float h2[256];     // [0,128)=corr branch, [128,256)=cat branch

  for (int j = tid; j < 600; j += 256) sbuf[j] = inA[(size_t)b * 600 + j];
  __syncthreads();
  {
    float s = b1a[tid];
    const float* wr = w1a + (size_t)tid * 600;
    for (int k = 0; k < 600; ++k) s += wr[k] * sbuf[k];
    h1[tid] = fmaxf(s, 0.f);
  }
  __syncthreads();
  if (tid < 128) {
    float s = b2a[tid];
    const float* wr = w2a + (size_t)tid * 256;
    for (int k = 0; k < 256; ++k) s += wr[k] * h1[k];
    h2[tid] = fmaxf(s, 0.f);
  }
  __syncthreads();
  for (int j = tid; j < 600; j += 256) sbuf[j] = inB[(size_t)b * 600 + j];
  __syncthreads();
  {
    float s = b1b[tid];
    const float* wr = w1b + (size_t)tid * 600;
    for (int k = 0; k < 600; ++k) s += wr[k] * sbuf[k];
    h1[tid] = fmaxf(s, 0.f);
  }
  __syncthreads();
  if (tid < 128) {
    float s = b2b[tid];
    const float* wr = w2b + (size_t)tid * 256;
    for (int k = 0; k < 256; ++k) s += wr[k] * h1[k];
    h2[128 + tid] = fmaxf(s, 0.f);
  }
  __syncthreads();
  if (tid < 2) {
    float s = bo[tid];
    const float* wr = wo + (size_t)tid * 256;
    for (int k = 0; k < 256; ++k) s += wr[k] * h2[k];
    out[(size_t)b * 2 + tid] = s;
  }
}

// ---------------------------------------------------------------------------
extern "C" void kernel_launch(void* const* d_in, const int* in_sizes, int n_in,
                              void* d_out, int out_size, void* d_ws, size_t ws_size,
                              hipStream_t stream) {
  const float* feat1 = (const float*)d_in[0];
  const float* feat2 = (const float*)d_in[1];
  const float* feat3 = (const float*)d_in[2];
  const float* Wa = (const float*)d_in[3];  const float* ba = (const float*)d_in[4];
  const float* Wb = (const float*)d_in[5];  const float* bb = (const float*)d_in[6];
  const float* Wc = (const float*)d_in[7];  const float* bc = (const float*)d_in[8];
  const float* Wcorr = (const float*)d_in[9];  const float* bcorr = (const float*)d_in[10];
  const float* Wcat  = (const float*)d_in[11]; const float* bcat  = (const float*)d_in[12];
  const float* cf_w1 = (const float*)d_in[13]; const float* cf_b1 = (const float*)d_in[14];
  const float* cf_w2 = (const float*)d_in[15]; const float* cf_b2 = (const float*)d_in[16];
  const float* ccf_w1 = (const float*)d_in[17]; const float* ccf_b1 = (const float*)d_in[18];
  const float* ccf_w2 = (const float*)d_in[19]; const float* ccf_b2 = (const float*)d_in[20];
  const float* Wout = (const float*)d_in[21]; const float* bout = (const float*)d_in[22];
  float* out = (float*)d_out;

  // workspace layout (bytes); X region is later reused for corrP (X dead then)
  char* ws = (char*)d_ws;
  const size_t szX    = (size_t)NPIX * Cc * 2;          // 26,214,400 per feat
  unsigned short* Xa = (unsigned short*)(ws);
  unsigned short* Xb = (unsigned short*)(ws + szX);
  unsigned short* Xc = (unsigned short*)(ws + 2 * szX);
  unsigned short* corrP = (unsigned short*)(ws);        // alias: needs 68.8MB <= 78.6MB
  const size_t corrBytes = (size_t)NPIX * CORR_PAD * 2;
  size_t off = 3 * szX;
  unsigned short* fcat = (unsigned short*)(ws + off); off += (size_t)NPIX * CAT_CH * 2;
  unsigned short* WaB = (unsigned short*)(ws + off); off += (size_t)Cc * Cc * 2;
  unsigned short* WbB = (unsigned short*)(ws + off); off += (size_t)Cc * Cc * 2;
  unsigned short* WcB = (unsigned short*)(ws + off); off += (size_t)Cc * Cc * 2;
  unsigned short* WcorrBT = (unsigned short*)(ws + off); off += (size_t)16 * 9 * CORR_PAD * 2;
  unsigned short* WcatBT  = (unsigned short*)(ws + off); off += (size_t)16 * 9 * CAT_CH * 2;
  float* cfA = (float*)(ws + off); off += (size_t)Bb * 600 * 4;
  float* cfB = (float*)(ws + off); off += (size_t)Bb * 600 * 4;
  (void)ws_size; (void)n_in; (void)in_sizes; (void)out_size;

  // K0: conversions
  k0_feat<<<dim3(2048, 1, 3), 256, 0, stream>>>(feat1, feat2, feat3, Xa, Xb, Xc);
  k0_w512<<<dim3(256, 1, 3), 256, 0, stream>>>(Wa, Wb, Wc, WaB, WbB, WcB);
  k0_wconv<<<dim3(128), 256, 0, stream>>>(Wcorr, WcorrBT, CORR_CH, CORR_PAD);
  k0_wconv<<<dim3(128), 256, 0, stream>>>(Wcat, WcatBT, CAT_CH, CAT_CH);

  // K1: three 1x1 convs -> fcat (channels-last bf16, concat for free)
  k1_conv1x1<<<dim3(NPIX / 64, Cc / 64, 3), 256, 0, stream>>>(
      Xa, Xb, Xc, WaB, WbB, WcB, ba, bb, bc, fcat);

  // zero the padded cost volume (invalid displacements + channel pad), then K2
  hipMemsetAsync(corrP, 0, corrBytes, stream);
  k2_corr<<<dim3(Hh, 3, Bb), 256, 0, stream>>>(fcat, corrP);

  // K3: corr 3x3 conv + relu -> [B,600];  K4: cat 3x3 conv -> [B,600]
  k34_conv3x3<<<dim3(NPIX / 128), 256, 0, stream>>>(corrP, CORR_PAD, WcorrBT,
                                                    bcorr, 1, cfA);
  k34_conv3x3<<<dim3(NPIX / 128), 256, 0, stream>>>(fcat, CAT_CH, WcatBT,
                                                    bcat, 0, cfB);

  // K5: MLPs + final linear -> [B,2]
  k5_mlp<<<dim3(Bb), 256, 0, stream>>>(cfA, cfB, cf_w1, cf_b1, cf_w2, cf_b2,
                                       ccf_w1, ccf_b1, ccf_w2, ccf_b2,
                                       Wout, bout, out);
}
